// CausalSelfAttention_1546188226781
// MI455X (gfx1250) — compile-verified, matrix operand order checked by reading, not run on hardware
//
#include <hip/hip_runtime.h>
#include <hip/hip_bf16.h>

// B=4, N=2048, D=1024, H=16, DH=64
// Pipeline:
//   convert_bf16   : x fp32 -> x_bf16 [8192,1024]
//   transpose_bf16 : w_qkv -> w_qkvT bf16 [3072,1024]; w_o -> w_oT bf16 [1024,1024]
//   gemm_wmma<0>   : qkv = x @ w_qkv + b_qkv, scatter q[B,H,N,DH], k[B,H,N,DH], vT[B,H,DH,N]
//   attn_wmma      : causal flash attention -> attn[B*N, D] bf16
//   gemm_wmma<1>   : out = attn @ w_o + b_o (fp32)
// GEMMs are LDS-free: operands laid out so a WMMA fragment is two contiguous
// 16B loads per lane, served from the 192MB L2 (weights/x stay resident).
// Softmax row reductions use DPP16 butterflies (pure VALU, co-executes with
// WMMA) instead of LDS bpermute.

typedef __bf16 bf16;
typedef __attribute__((ext_vector_type(4)))  __bf16 v4bf;
typedef __attribute__((ext_vector_type(8)))  __bf16 v8bf;
typedef __attribute__((ext_vector_type(16))) __bf16 v16bf;
typedef __attribute__((ext_vector_type(8)))  float  v8f;
typedef __attribute__((ext_vector_type(4)))  float  v4f;

__device__ __forceinline__ v8f wmma_bf16(v16bf a, v16bf b, v8f c) {
  return __builtin_amdgcn_wmma_f32_16x16x32_bf16(false, a, false, b, (short)0, c,
                                                 false, false);
}

// Per-lane data for a 16x32 bf16 WMMA operand: two contiguous 16B chunks at
// +0 and +16 elements (lane holds row lane&15; K = (lane>>4)*8 + {0..7,16..23}).
__device__ __forceinline__ v16bf ld_frag(const bf16* p) {
  v8bf lo = *(const v8bf*)p;
  v8bf hi = *(const v8bf*)(p + 16);
  return __builtin_shufflevector(lo, hi, 0, 1, 2, 3, 4, 5, 6, 7, 8, 9, 10, 11,
                                 12, 13, 14, 15);
}

// ---- DPP16 16-lane butterfly reductions (VALU only, both wave halves) ----
// DPP ctrl must be an immediate -> template parameter.
template <int CTRL>
__device__ __forceinline__ float dpp16(float x) {
  return __int_as_float(
      __builtin_amdgcn_update_dpp(0, __float_as_int(x), CTRL, 0xF, 0xF, true));
}
__device__ __forceinline__ float red16_max(float x) {
  x = fmaxf(x, dpp16<0xB1>(x));   // quad_perm [1,0,3,2]
  x = fmaxf(x, dpp16<0x4E>(x));   // quad_perm [2,3,0,1]
  x = fmaxf(x, dpp16<0x141>(x));  // row_half_mirror
  x = fmaxf(x, dpp16<0x140>(x));  // row_mirror
  return x;
}
__device__ __forceinline__ float red16_sum(float x) {
  x += dpp16<0xB1>(x);
  x += dpp16<0x4E>(x);
  x += dpp16<0x141>(x);
  x += dpp16<0x140>(x);
  return x;
}

// ---------------------------------------------------------------------------
// Elementwise fp32 -> bf16 (vector x4)
// ---------------------------------------------------------------------------
__global__ __launch_bounds__(256) void convert_bf16(const float* __restrict__ src,
                                                    bf16* __restrict__ dst,
                                                    long n4) {
  long i = ((long)blockIdx.x * 256 + threadIdx.x) * 4;
  v4f v = *(const v4f*)(src + i);
  v4bf o;
#pragma unroll
  for (int j = 0; j < 4; ++j) o[j] = (bf16)v[j];
  *(v4bf*)(dst + i) = o;
}

// ---------------------------------------------------------------------------
// src [R][C] fp32 -> dst [C][R] bf16, 32x32 LDS tiles, 256 threads
// ---------------------------------------------------------------------------
__global__ __launch_bounds__(256) void transpose_bf16(const float* __restrict__ src,
                                                      bf16* __restrict__ dst,
                                                      int R, int C) {
  __shared__ float tile[32][33];
  const int tc = blockIdx.x * 32;
  const int tr = blockIdx.y * 32;
  const int lx = threadIdx.x & 31;
  const int ly = threadIdx.x >> 5;  // 0..7
#pragma unroll
  for (int i = 0; i < 32; i += 8)
    tile[ly + i][lx] = src[(long)(tr + ly + i) * C + tc + lx];
  __syncthreads();
#pragma unroll
  for (int i = 0; i < 32; i += 8)
    dst[(long)(tc + ly + i) * R + tr + lx] = (bf16)tile[lx][ly + i];
}

// ---------------------------------------------------------------------------
// GEMM: C[M,N] = A[M,K] @ B + bias, B given transposed: BT[N][K] bf16.
// LDS-free. Block tile 128x256 = 8 waves (2x4) of 64x64 wave tiles.
// MODE 0: QKV scatter epilogue (bf16), MODE 1: plain fp32 store.
// Epilogue decode (q/k/v select, head, batch) is wave-uniform by construction
// (16-wide N tiles never straddle the 1024 q/k/v boundary; 128-row M tiles
// never straddle the 2048 batch boundary) -> scalar branches only.
// ---------------------------------------------------------------------------
template <int MODE>
__global__ __launch_bounds__(256) void gemm_wmma(
    const bf16* __restrict__ A, const bf16* __restrict__ BT,
    const float* __restrict__ bias, float* __restrict__ Cout,
    bf16* __restrict__ q_out, bf16* __restrict__ k_out,
    bf16* __restrict__ vT_out, int M, int N, int K) {
  const int tid = threadIdx.x;
  const int lane = tid & 31, wave = tid >> 5;
  const int lmod = lane & 15, lhalf = lane >> 4;
  const int kb8 = lhalf * 8;
  const int wrow = wave >> 2, wcol = wave & 3;  // 2x4 wave grid
  const int mW = blockIdx.y * 128 + wrow * 64;
  const int nW = blockIdx.x * 256 + wcol * 64;

  const bf16* aRow[4];
  const bf16* bRow[4];
#pragma unroll
  for (int t = 0; t < 4; ++t) {
    aRow[t] = A + (long)(mW + t * 16 + lmod) * K + kb8;
    bRow[t] = BT + (long)(nW + t * 16 + lmod) * K + kb8;
  }

  v8f acc[4][4];
#pragma unroll
  for (int mt = 0; mt < 4; ++mt)
#pragma unroll
    for (int nt = 0; nt < 4; ++nt)
#pragma unroll
      for (int r = 0; r < 8; ++r) acc[mt][nt][r] = 0.0f;

  for (int k0 = 0; k0 < K; k0 += 32) {
    v16bf af[4], bfr[4];
#pragma unroll
    for (int t = 0; t < 4; ++t) af[t] = ld_frag(aRow[t] + k0);
#pragma unroll
    for (int t = 0; t < 4; ++t) bfr[t] = ld_frag(bRow[t] + k0);
    if (k0 + 32 < K) {  // prefetch next K-step (global_prefetch_b8)
#pragma unroll
      for (int t = 0; t < 4; ++t) {
        __builtin_prefetch(aRow[t] + k0 + 32, 0, 3);
        __builtin_prefetch(bRow[t] + k0 + 32, 0, 3);
      }
    }
#pragma unroll
    for (int mt = 0; mt < 4; ++mt)
#pragma unroll
      for (int nt = 0; nt < 4; ++nt)
        acc[mt][nt] = wmma_bf16(af[mt], bfr[nt], acc[mt][nt]);
  }

  // ---- epilogue ----
  const int bb = mW >> 11;       // batch (block-uniform)
  const int nnBase = mW & 2047;  // seq base (wave-uniform)
#pragma unroll
  for (int nt = 0; nt < 4; ++nt) {
    const int nTile = nW + nt * 16;  // wave-uniform
    const float bv = bias[nTile + lmod];
    if (MODE == 1) {
#pragma unroll
      for (int mt = 0; mt < 4; ++mt) {
        float* dst = Cout + (long)(mW + mt * 16 + 8 * lhalf) * N + nTile + lmod;
#pragma unroll
        for (int r = 0; r < 8; ++r) dst[(long)r * N] = acc[mt][nt][r] + bv;
      }
    } else {
      const int which = nTile >> 10;  // 0=q 1=k 2=v (wave-uniform)
      const int hn = nTile & 1023;
      const int h = hn >> 6;          // head (wave-uniform)
      const int dhB = hn & 63;        // head-dim base (wave-uniform)
      const long headBase = (long)(bb * 16 + h);
      if (which == 2) {
        // vT[B,H,DH,N]: consecutive r adjacent -> one b128 store per tile
#pragma unroll
        for (int mt = 0; mt < 4; ++mt) {
          bf16* dst = vT_out + (headBase * 64 + dhB + lmod) * 2048 + nnBase +
                      mt * 16 + 8 * lhalf;
          v8bf o;
#pragma unroll
          for (int r = 0; r < 8; ++r) o[r] = (bf16)(acc[mt][nt][r] + bv);
          *(v8bf*)dst = o;
        }
      } else {
        bf16* base = (which == 0 ? q_out : k_out) +
                     (headBase * 2048 + nnBase) * 64 + dhB + lmod;
#pragma unroll
        for (int mt = 0; mt < 4; ++mt) {
          bf16* dst = base + (long)(mt * 16 + 8 * lhalf) * 64;
#pragma unroll
          for (int r = 0; r < 8; ++r)
            dst[(long)r * 64] = (bf16)(acc[mt][nt][r] + bv);
        }
      }
    }
  }
}

// ---------------------------------------------------------------------------
// Causal flash attention, scale = 1/sqrt(64). Each wave independently owns
// one (b,h, 16-query) tile (no block barriers). V fragments load BEFORE the
// softmax so its VALU work hides their latency; DPP16 butterflies do the row
// reductions on the VALU pipe; next key-block K rows are prefetched.
// ---------------------------------------------------------------------------
__global__ __launch_bounds__(256) void attn_wmma(const bf16* __restrict__ q,
                                                 const bf16* __restrict__ k,
                                                 const bf16* __restrict__ vT,
                                                 bf16* __restrict__ attn_out) {
  __shared__ __align__(16) bf16 sP[8][16][64];  // per-wave P tile (16q x 64k)

  const int tid = threadIdx.x;
  const int lane = tid & 31, wave = tid >> 5;
  const int lmod = lane & 15, lhalf = lane >> 4;
  const int kb8 = lhalf * 8;

  const int wid = blockIdx.x * 8 + wave;  // 0..8191
  const int bh = wid >> 7;                // b*16+h
  const int qbase = (wid & 127) * 16;
  const float scale = 0.125f;             // 1/sqrt(64)

  const bf16* qrow = q + ((long)bh * 2048 + qbase + lmod) * 64;
  v16bf aq[2];
#pragma unroll
  for (int kc = 0; kc < 2; ++kc) aq[kc] = ld_frag(qrow + kc * 32 + kb8);

  v8f o[4];
  float m_i[8], l_i[8];
#pragma unroll
  for (int nt = 0; nt < 4; ++nt)
#pragma unroll
    for (int r = 0; r < 8; ++r) o[nt][r] = 0.0f;
#pragma unroll
  for (int r = 0; r < 8; ++r) {
    m_i[r] = -3.0e38f;
    l_i[r] = 0.0f;
  }

  const int nkb = (qbase + 16 + 63) >> 6;  // causal key-block count
  for (int kb = 0; kb < nkb; ++kb) {
    const int kstart = kb * 64;

    // ---- S = Q @ K^T (16 x 64) ----
    v8f s[4];
#pragma unroll
    for (int nt = 0; nt < 4; ++nt) {
#pragma unroll
      for (int r = 0; r < 8; ++r) s[nt][r] = 0.0f;
#pragma unroll
      for (int kc = 0; kc < 2; ++kc) {
        const bf16* krow = k + ((long)bh * 2048 + kstart + nt * 16 + lmod) * 64 +
                           kc * 32 + kb8;
        s[nt] = wmma_bf16(aq[kc], ld_frag(krow), s[nt]);
      }
    }

    // ---- hoist V fragment loads; softmax below hides their latency ----
    v16bf vf[4][2];
#pragma unroll
    for (int nt = 0; nt < 4; ++nt)
#pragma unroll
      for (int kc = 0; kc < 2; ++kc)
        vf[nt][kc] = ld_frag(vT + ((long)bh * 64 + nt * 16 + lmod) * 2048 +
                             kstart + kc * 32 + kb8);

    if (kstart + 64 < 2048) {  // prefetch next key block (global_prefetch_b8)
#pragma unroll
      for (int nt = 0; nt < 4; ++nt)
        __builtin_prefetch(
            k + ((long)bh * 2048 + kstart + 64 + nt * 16 + lmod) * 64 + kb8, 0, 3);
    }

    // ---- causal mask + online softmax (rows live in 16-lane halves) ----
#pragma unroll
    for (int r = 0; r < 8; ++r) {
      const int gq = qbase + 8 * lhalf + r;
      float sv[4];
      float mx = -3.0e38f;
#pragma unroll
      for (int nt = 0; nt < 4; ++nt) {
        const int gk = kstart + nt * 16 + lmod;
        float v = s[nt][r] * scale;
        if (gk > gq) v = -3.0e38f;
        sv[nt] = v;
        mx = fmaxf(mx, v);
      }
      mx = red16_max(mx);  // DPP butterfly, VALU only
      const float mnew = fmaxf(m_i[r], mx);
      const float alpha = __expf(m_i[r] - mnew);
      float rs = 0.0f;
#pragma unroll
      for (int nt = 0; nt < 4; ++nt) {
        const float p = __expf(sv[nt] - mnew);
        rs += p;
        sP[wave][8 * lhalf + r][nt * 16 + lmod] = (bf16)p;  // C->A reshape
      }
      rs = red16_sum(rs);  // DPP butterfly, VALU only
      l_i[r] = l_i[r] * alpha + rs;
      m_i[r] = mnew;
#pragma unroll
      for (int nt = 0; nt < 4; ++nt) o[nt][r] *= alpha;
    }

    // ---- P fragments from wave-private LDS (DScnt orders the RAW) ----
    v16bf ap[2];
#pragma unroll
    for (int kc = 0; kc < 2; ++kc)
      ap[kc] = ld_frag(&sP[wave][lmod][kc * 32 + kb8]);

    // ---- O += P @ V ----
#pragma unroll
    for (int nt = 0; nt < 4; ++nt)
#pragma unroll
      for (int kc = 0; kc < 2; ++kc)
        o[nt] = wmma_bf16(ap[kc], vf[nt][kc], o[nt]);
  }

  // ---- finalize: divide by row sum, store attn[B*N, H*DH] bf16 ----
  const int b = bh >> 4, h = bh & 15;
#pragma unroll
  for (int r = 0; r < 8; ++r) {
    const float inv = 1.0f / l_i[r];
    const int gq = qbase + 8 * lhalf + r;
#pragma unroll
    for (int nt = 0; nt < 4; ++nt) {
      const int dh = nt * 16 + lmod;
      attn_out[((long)b * 2048 + gq) * 1024 + h * 64 + dh] =
          (bf16)(o[nt][r] * inv);
    }
  }
}

// ---------------------------------------------------------------------------
extern "C" void kernel_launch(void* const* d_in, const int* in_sizes, int n_in,
                              void* d_out, int out_size, void* d_ws,
                              size_t ws_size, hipStream_t stream) {
  const float* x = (const float*)d_in[0];      // [4,2048,1024]
  const float* w_qkv = (const float*)d_in[1];  // [1024,3072]
  const float* b_qkv = (const float*)d_in[2];  // [3072]
  const float* w_o = (const float*)d_in[3];    // [1024,1024]
  const float* b_o = (const float*)d_in[4];    // [1024]
  float* out = (float*)d_out;                  // [4,2048,1024]

  const long E = 8388608L;       // 4*2048*1024 elements
  bf16* qb = (bf16*)d_ws;        // 16 MB
  bf16* kbuf = qb + E;           // 16 MB
  bf16* vTb = kbuf + E;          // 16 MB
  bf16* attn = vTb + E;          // 16 MB
  bf16* xbf = attn + E;          // 16 MB
  bf16* wqkvT = xbf + E;         // [3072,1024] 6 MB
  bf16* woT = wqkvT + 3145728L;  // [1024,1024] 2 MB   (total ~88 MB)

  convert_bf16<<<dim3(8192), 256, 0, stream>>>(x, xbf, E / 4);
  transpose_bf16<<<dim3(96, 32), 256, 0, stream>>>(w_qkv, wqkvT, 1024, 3072);
  transpose_bf16<<<dim3(32, 32), 256, 0, stream>>>(w_o, woT, 1024, 1024);

  gemm_wmma<0><<<dim3(12, 64), 256, 0, stream>>>(
      xbf, wqkvT, b_qkv, nullptr, qb, kbuf, vTb, 8192, 3072, 1024);

  attn_wmma<<<dim3(1024), 256, 0, stream>>>(qb, kbuf, vTb, attn);

  gemm_wmma<1><<<dim3(4, 64), 256, 0, stream>>>(
      attn, woT, b_o, out, nullptr, nullptr, nullptr, 8192, 1024, 1024);
}
